// GraphAttentionEncoder_82592221102836
// MI455X (gfx1250) — compile-verified, matrix operand order checked by reading, not run on hardware
//
#include <hip/hip_runtime.h>
#include <float.h>

typedef __attribute__((ext_vector_type(16))) _Float16 v16h;
typedef __attribute__((ext_vector_type(8)))  float    v8f;

#define LRELU_NEG 0.2f
#define LN_EPS    1e-5f

static __device__ __forceinline__ float lrelu(float x) { return x > 0.0f ? x : LRELU_NEG * x; }

// order-preserving float -> uint encoding (for atomicMax on floats)
static __device__ __forceinline__ unsigned fenc(float f) {
  unsigned u = __float_as_uint(f);
  return (u & 0x80000000u) ? ~u : (u | 0x80000000u);
}
static __device__ __forceinline__ float fdec(unsigned u) {
  return __uint_as_float((u & 0x80000000u) ? (u & 0x7fffffffu) : ~u);
}
static __device__ __forceinline__ void atomAddF(float* p, float v) {
  __hip_atomic_fetch_add(p, v, __ATOMIC_RELAXED, __HIP_MEMORY_SCOPE_AGENT);
}

// ---------------------------------------------------------------- f32 -> f16
__global__ __launch_bounds__(256) void cvt_f32_f16(const float* __restrict__ src,
                                                   _Float16* __restrict__ dst, int n) {
  int i = blockIdx.x * blockDim.x + threadIdx.x;
  if (i < n) dst[i] = (_Float16)src[i];
}

// ---------------------------------------------------------------- WMMA GEMM
// Hout[n][o] = sum_k Xh[n][k] * Wh[o][k]   (Xh: f16 [n,128], Wh: f16 [128,128])
// Block: 256 threads = 8 waves; wave w owns output tile rows[blk*16..+16) x cols[w*16..+16)
__global__ __launch_bounds__(256) void gat_gemm_wmma(const _Float16* __restrict__ Xh,
                                                     const _Float16* __restrict__ Wh,
                                                     float* __restrict__ Hout, int n) {
  const int wave = threadIdx.x >> 5;
  const int lane = threadIdx.x & 31;
  const int hs   = lane >> 4;      // which K-half this lane holds
  const int m    = lane & 15;      // A row / B column within tile
  const int row_base = blockIdx.x << 4;
  const int col_base = wave << 4;
  int arow = row_base + m;
  if (arow >= n) arow = n - 1;     // clamp (harmless duplicate reads)
  const _Float16* xr = Xh + (size_t)arow * 128;
  const _Float16* wr = Wh + (size_t)(col_base + m) * 128;
  // prefetch next row-block's A tile (global_prefetch_b8)
  if (row_base + 16 + m < n) __builtin_prefetch(xr + 128 * 16, 0, 1);
  v8f acc = {};
#pragma unroll
  for (int k0 = 0; k0 < 128; k0 += 32) {
    const int ka = k0 + hs * 8;    // lanes 0-15: K={0..7,16..23}; lanes 16-31: K={8..15,24..31}
    v16h a, b;
#pragma unroll
    for (int j = 0; j < 8; ++j) {
      a[j]     = xr[ka + j];
      a[j + 8] = xr[ka + 16 + j];
      b[j]     = wr[ka + j];
      b[j + 8] = wr[ka + 16 + j];
    }
    acc = __builtin_amdgcn_wmma_f32_16x16x32_f16(false, a, false, b, (short)0, acc,
                                                 false, false);
  }
  // D layout: VGPR r, lane L -> row = r + (L>>4)*8, col = L&15
  float* outp = Hout + (size_t)(row_base + hs * 8) * 128 + col_base + m;
  if (row_base + 16 <= n) {        // uniform (scalar) branch: full tile in bounds
#pragma unroll
    for (int r = 0; r < 8; ++r) outp[(size_t)r * 128] = acc[r];
  } else {                         // tail block only
#pragma unroll
    for (int r = 0; r < 8; ++r)
      if (row_base + hs * 8 + r < n) outp[(size_t)r * 128] = acc[r];
  }
}

// ---------------------------------------------------------------- per-layer init
__global__ __launch_bounds__(256) void gat_init(unsigned* __restrict__ mmax,
                                                float* __restrict__ denom,
                                                float* __restrict__ agg, int n, int Hh) {
  int t = blockIdx.x * blockDim.x + threadIdx.x;
  if (t < n * 128) agg[t] = 0.0f;
  if (t < n * Hh) { mmax[t] = 0u; denom[t] = 0.0f; }   // 0u < fenc(any finite float)
}

// ---------------------------------------------------------------- node scores
__global__ __launch_bounds__(256) void gat_node_scores(const float* __restrict__ Hf,
                                                       const float* __restrict__ a_s,
                                                       const float* __restrict__ a_d,
                                                       float* __restrict__ es,
                                                       float* __restrict__ ed,
                                                       int n, int Hh, int C) {
  int t = blockIdx.x * blockDim.x + threadIdx.x;
  if (t >= n * Hh) return;
  int node = t / Hh, head = t - node * Hh;
  const float* hp = Hf + (size_t)node * 128 + head * C;
  const float* as = a_s + head * C;
  const float* ad = a_d + head * C;
  float s = 0.0f, d = 0.0f;
  for (int c = 0; c < C; ++c) { float v = hp[c]; s += v * as[c]; d += v * ad[c]; }
  es[t] = s; ed[t] = d;
}

// ---------------------------------------------------------------- edge pass 1: segment max
__global__ __launch_bounds__(256) void gat_edge_max(const int* __restrict__ ei, int E, int n,
                                                    const float* __restrict__ es,
                                                    const float* __restrict__ ed,
                                                    unsigned* __restrict__ mmax, int Hh) {
  int t = blockIdx.x * blockDim.x + threadIdx.x;
  if (t >= (E + n) * Hh) return;
  int e = t / Hh, head = t - e * Hh;
  int s, d;
  if (e < E) { s = ei[e]; d = ei[E + e]; } else { s = d = e - E; }
  float ev = lrelu(es[s * Hh + head] + ed[d * Hh + head]);
  atomicMax(&mmax[d * Hh + head], fenc(ev));
}

// ---------------------------------------------------------------- edge pass 2: exp + denom
__global__ __launch_bounds__(256) void gat_edge_w(const int* __restrict__ ei, int E, int n,
                                                  const float* __restrict__ es,
                                                  const float* __restrict__ ed,
                                                  const unsigned* __restrict__ mmax,
                                                  float* __restrict__ wbuf,
                                                  float* __restrict__ denom, int Hh) {
  int t = blockIdx.x * blockDim.x + threadIdx.x;
  if (t >= (E + n) * Hh) return;
  int e = t / Hh, head = t - e * Hh;
  int s, d;
  if (e < E) { s = ei[e]; d = ei[E + e]; } else { s = d = e - E; }
  float ev = lrelu(es[s * Hh + head] + ed[d * Hh + head]);
  float wv = __expf(ev - fdec(mmax[d * Hh + head]));
  wbuf[t] = wv;
  atomAddF(&denom[d * Hh + head], wv);
}

// ---------------------------------------------------------------- edge pass 3: aggregation
// wave per edge; lane handles 4 channels (head = c / C is uniform within the float4)
__global__ __launch_bounds__(256) void gat_edge_agg(const int* __restrict__ ei, int E, int n,
                                                    const float* __restrict__ Hf,
                                                    const float* __restrict__ wbuf,
                                                    const float* __restrict__ denom,
                                                    float* __restrict__ agg, int Hh, int C) {
  int t = blockIdx.x * blockDim.x + threadIdx.x;
  int e = t >> 5;
  if (e >= E + n) return;
  int lane = t & 31;
  int c = lane << 2;
  int head = c / C;
  int s, d;
  if (e < E) { s = ei[e]; d = ei[E + e]; } else { s = d = e - E; }
  float alpha = wbuf[e * Hh + head] / denom[d * Hh + head];
  const float* hp = Hf + (size_t)s * 128 + c;
  float* op = agg + (size_t)d * 128 + c;
  atomAddF(op + 0, alpha * hp[0]);
  atomAddF(op + 1, alpha * hp[1]);
  atomAddF(op + 2, alpha * hp[2]);
  atomAddF(op + 3, alpha * hp[3]);
}

// ---------------------------------------------------------------- epilogue: +b, LN, act, +res
// wave per node; wave32 shuffle reductions across 128 features
__global__ __launch_bounds__(256) void gat_epilogue(const float* __restrict__ agg,
                                                    const float* __restrict__ bias,
                                                    const float* __restrict__ gamma,
                                                    const float* __restrict__ beta,
                                                    const float* __restrict__ res,
                                                    float* __restrict__ xout, int n, int act) {
  int t = blockIdx.x * blockDim.x + threadIdx.x;
  int node = t >> 5;
  if (node >= n) return;
  int lane = t & 31;
  int c = lane << 2;
  const float* ap = agg + (size_t)node * 128 + c;
  float v0 = ap[0] + bias[c + 0];
  float v1 = ap[1] + bias[c + 1];
  float v2 = ap[2] + bias[c + 2];
  float v3 = ap[3] + bias[c + 3];
  float s = v0 + v1 + v2 + v3;
  for (int off = 16; off > 0; off >>= 1) s += __shfl_xor(s, off, 32);
  float mu = s * (1.0f / 128.0f);
  float d0 = v0 - mu, d1 = v1 - mu, d2 = v2 - mu, d3 = v3 - mu;
  float q = d0 * d0 + d1 * d1 + d2 * d2 + d3 * d3;
  for (int off = 16; off > 0; off >>= 1) q += __shfl_xor(q, off, 32);
  float inv = rsqrtf(q * (1.0f / 128.0f) + LN_EPS);
  float o0 = d0 * inv * gamma[c + 0] + beta[c + 0];
  float o1 = d1 * inv * gamma[c + 1] + beta[c + 1];
  float o2 = d2 * inv * gamma[c + 2] + beta[c + 2];
  float o3 = d3 * inv * gamma[c + 3] + beta[c + 3];
  if (act) {
    const float* rp = res + (size_t)node * 128 + c;
    o0 = lrelu(o0) + rp[0];
    o1 = lrelu(o1) + rp[1];
    o2 = lrelu(o2) + rp[2];
    o3 = lrelu(o3) + rp[3];
  }
  float* xo = xout + (size_t)node * 128 + c;
  xo[0] = o0; xo[1] = o1; xo[2] = o2; xo[3] = o3;
}

// ---------------------------------------------------------------- pooling
__global__ __launch_bounds__(256) void pool_init(unsigned* __restrict__ cnt,
                                                 float* __restrict__ macc,
                                                 unsigned* __restrict__ xacc, int G) {
  int t = blockIdx.x * blockDim.x + threadIdx.x;
  if (t < G * 128) { macc[t] = 0.0f; xacc[t] = fenc(-FLT_MAX); }
  if (t < G) cnt[t] = 0u;
}

__global__ __launch_bounds__(256) void pool_scatter(const float* __restrict__ X,
                                                    const int* __restrict__ batch,
                                                    unsigned* __restrict__ cnt,
                                                    float* __restrict__ macc,
                                                    unsigned* __restrict__ xacc, int n) {
  int t = blockIdx.x * blockDim.x + threadIdx.x;
  int node = t >> 5;
  if (node >= n) return;
  int lane = t & 31;
  int c = lane << 2;
  int g = batch[node];
  const float* xp = X + (size_t)node * 128 + c;
  float* mp = macc + (size_t)g * 128 + c;
  unsigned* xa = xacc + (size_t)g * 128 + c;
  atomAddF(mp + 0, xp[0]); atomicMax(xa + 0, fenc(xp[0]));
  atomAddF(mp + 1, xp[1]); atomicMax(xa + 1, fenc(xp[1]));
  atomAddF(mp + 2, xp[2]); atomicMax(xa + 2, fenc(xp[2]));
  atomAddF(mp + 3, xp[3]); atomicMax(xa + 3, fenc(xp[3]));
  if (lane == 0) atomicAdd(&cnt[g], 1u);
}

__global__ __launch_bounds__(256) void pool_final(const unsigned* __restrict__ cnt,
                                                  const float* __restrict__ macc,
                                                  const unsigned* __restrict__ xacc,
                                                  float* __restrict__ out, int G) {
  int t = blockIdx.x * blockDim.x + threadIdx.x;
  if (t >= G * 128) return;
  int g = t >> 7, c = t & 127;
  float cn = (float)cnt[g];
  if (cn < 1.0f) cn = 1.0f;
  out[(size_t)g * 256 + c]       = macc[t] / cn;
  out[(size_t)g * 256 + 128 + c] = fdec(xacc[t]);
}

// ================================================================ host
extern "C" void kernel_launch(void* const* d_in, const int* in_sizes, int n_in,
                              void* d_out, int out_size, void* d_ws, size_t ws_size,
                              hipStream_t stream) {
  const float* x     = (const float*)d_in[0];
  const int*   ei    = (const int*)d_in[1];
  const int*   batch = (const int*)d_in[2];
  const float* Wf[3] = {(const float*)d_in[3], (const float*)d_in[9],  (const float*)d_in[15]};
  const float* As[3] = {(const float*)d_in[4], (const float*)d_in[10], (const float*)d_in[16]};
  const float* Ad[3] = {(const float*)d_in[5], (const float*)d_in[11], (const float*)d_in[17]};
  const float* Bb[3] = {(const float*)d_in[6], (const float*)d_in[12], (const float*)d_in[18]};
  const float* Gg[3] = {(const float*)d_in[7], (const float*)d_in[13], (const float*)d_in[19]};
  const float* Be[3] = {(const float*)d_in[8], (const float*)d_in[14], (const float*)d_in[20]};

  const int N  = in_sizes[0] / 128;
  const int E  = in_sizes[1] / 2;
  const int G  = out_size / 256;
  const int ET = E + N;

  // workspace carve (256B aligned)
  char* ws = (char*)d_ws;
  size_t off = 0;
  auto carve = [&](size_t bytes) {
    char* p = ws + off;
    off += (bytes + 255) & ~(size_t)255;
    return (void*)p;
  };
  float*    bufA  = (float*)carve((size_t)N * 128 * 4);
  float*    bufB  = (float*)carve((size_t)N * 128 * 4);
  float*    bufC  = (float*)carve((size_t)N * 128 * 4);
  _Float16* Xh    = (_Float16*)carve((size_t)N * 128 * 2);
  float*    es    = (float*)carve((size_t)N * 8 * 4);
  float*    ed    = (float*)carve((size_t)N * 8 * 4);
  unsigned* mmax  = (unsigned*)carve((size_t)N * 8 * 4);
  float*    denom = (float*)carve((size_t)N * 8 * 4);
  float*    wbuf  = (float*)carve((size_t)ET * 8 * 4);
  _Float16* Wh    = (_Float16*)carve(128 * 128 * 2);
  unsigned* cnt   = (unsigned*)carve((size_t)G * 4);
  float*    macc  = (float*)carve((size_t)G * 128 * 4);
  unsigned* xacc  = (unsigned*)carve((size_t)G * 128 * 4);
  (void)ws_size;

  const int TB = 256;
  auto blocks = [&](long long work) { return (int)((work + TB - 1) / TB); };

  // layer plumbing: xin, residual, h-buffer, agg-buffer, xout, heads, channels
  struct Layer { const float* xin; const float* res; float* hb; float* ob; float* xo; int Hh, C; };
  Layer layers[3] = {
      {x,    x,    bufA, bufB, bufC, 8, 16},
      {bufC, bufC, bufA, bufB, bufA, 8, 16},
      {bufA, 0,    bufC, bufB, bufC, 1, 128},
  };

  for (int l = 0; l < 3; ++l) {
    const Layer& L = layers[l];
    cvt_f32_f16<<<blocks(128 * 128), TB, 0, stream>>>(Wf[l], Wh, 128 * 128);
    cvt_f32_f16<<<blocks((long long)N * 128), TB, 0, stream>>>(L.xin, Xh, N * 128);
    gat_gemm_wmma<<<(N + 15) / 16, TB, 0, stream>>>(Xh, Wh, L.hb, N);
    gat_init<<<blocks((long long)N * 128), TB, 0, stream>>>(mmax, denom, L.ob, N, L.Hh);
    gat_node_scores<<<blocks((long long)N * L.Hh), TB, 0, stream>>>(L.hb, As[l], Ad[l], es, ed, N,
                                                                    L.Hh, L.C);
    gat_edge_max<<<blocks((long long)ET * L.Hh), TB, 0, stream>>>(ei, E, N, es, ed, mmax, L.Hh);
    gat_edge_w<<<blocks((long long)ET * L.Hh), TB, 0, stream>>>(ei, E, N, es, ed, mmax, wbuf,
                                                                denom, L.Hh);
    gat_edge_agg<<<blocks((long long)ET * 32), TB, 0, stream>>>(ei, E, N, L.hb, wbuf, denom, L.ob,
                                                                L.Hh, L.C);
    gat_epilogue<<<blocks((long long)N * 32), TB, 0, stream>>>(L.ob, Bb[l], Gg[l], Be[l], L.res,
                                                               L.xo, N, l < 2 ? 1 : 0);
  }

  pool_init<<<blocks((long long)G * 128), TB, 0, stream>>>(cnt, macc, xacc, G);
  pool_scatter<<<blocks((long long)N * 32), TB, 0, stream>>>(bufC, batch, cnt, macc, xacc, N);
  pool_final<<<blocks((long long)G * 128), TB, 0, stream>>>(cnt, macc, xacc, (float*)d_out, G);
}
